// GatedDeltaNet_78855599554936
// MI455X (gfx1250) — compile-verified, matrix operand order checked by reading, not run on hardware
//
#include <hip/hip_runtime.h>
#include <math.h>

// GatedDeltaNet forward for MI455X (gfx1250, wave32).
// B=4 S=4096 HID=1024 H=16 D=64 K=4 (derived from in_sizes at launch).
//
// Pipeline:
//   1. one-time f32->bf16 conversion of x and all GEMM weights (halves GEMM read traffic)
//   2. bf16 WMMA GEMMs (q/k/v raw, alpha/beta sigmoid): 64x128 block tile, 4 waves,
//      each wave 32x64 (2x4 WMMA accs), LDS double-buffered via
//      global_load_async_to_lds_b128 + s_wait_asynccnt (ASYNCcnt pipeline)
//   3. depthwise causal conv(K=4) + SiLU
//   4. wave32 delta recurrence: state in 128 VGPRs/lane, shuffle-only reductions,
//      alpha gate + GroupNorm fused, emits bf16 directly
//   5. bf16 WMMA output projection

#define H_    16
#define D_    64
#define HID_  1024
#define TOT_  1024
#define KCONV 4

typedef __attribute__((ext_vector_type(16))) __bf16 v16bf;
typedef __attribute__((ext_vector_type(8)))  float  v8f;

struct U8x4 { unsigned u[8]; };

__device__ __forceinline__ unsigned short f2bf(float f) {
  union { float f; unsigned u; } v; v.f = f;
  unsigned r = v.u + 0x7FFFu + ((v.u >> 16) & 1u);  // round-to-nearest-even
  return (unsigned short)(r >> 16);
}

// -------- CDNA5 async global->LDS (ASYNCcnt) --------
__device__ __forceinline__ void async_copy_b128(unsigned lds_byte_off, const void* gaddr) {
  asm volatile("global_load_async_to_lds_b128 %0, %1, off"
               :: "v"(lds_byte_off), "v"(gaddr) : "memory");
}
__device__ __forceinline__ void wait_async_le6() { asm volatile("s_wait_asynccnt 0x6" ::: "memory"); }
__device__ __forceinline__ void wait_async_le0() { asm volatile("s_wait_asynccnt 0x0" ::: "memory"); }

// Low 32 bits of a generic LDS pointer == LDS byte offset (aperture mapping).
__device__ __forceinline__ unsigned lds_offset_of(const void* p) {
  return (unsigned)(unsigned long long)p;
}

// Gather one 16x32 bf16 WMMA operand fragment from an LDS tile stored [row][k]
// (row stride 32 bf16). CDNA5 ISA 7.12.2 layout: lanes 0-15 -> M/N=lane,
// K {0..7,16..23}; lanes 16-31 -> K {8..15,24..31}; each VGPR = (K,K+1) pair.
// Maps to two ds_load_b128 per fragment.
__device__ __forceinline__ v16bf frag_from_lds(const unsigned short* Ls, int row16, int lane) {
  const int r  = row16 + (lane & 15);
  const int kb = (lane >> 4) << 3;  // 0 or 8
  U8x4 t;
#pragma unroll
  for (int i = 0; i < 8; ++i) {
    const int k0 = ((i < 4) ? 0 : 16) + kb + ((i & 3) << 1);
    t.u[i] = *(const unsigned*)(&Ls[r * 32 + k0]);
  }
  return __builtin_bit_cast(v16bf, t);
}

// -------- f32 -> bf16 conversion pass --------
__global__ __launch_bounds__(256)
void cvt_bf16_kernel(const float* __restrict__ s, unsigned short* __restrict__ d, size_t n) {
  const size_t i = ((size_t)blockIdx.x * 256 + threadIdx.x) * 4;
  if (i + 3 < n) {
    const float4 f = *(const float4*)(s + i);
    uint2 o;
    o.x = (unsigned)f2bf(f.x) | ((unsigned)f2bf(f.y) << 16);
    o.y = (unsigned)f2bf(f.z) | ((unsigned)f2bf(f.w) << 16);
    *(uint2*)(d + i) = o;
  } else {
    for (size_t j = i; j < n; ++j) d[j] = f2bf(s[j]);
  }
}

// -------- bf16 WMMA GEMM: Y[m,n] = sum_k A[m,k]*W[n,k] (+bias) (mode1: sigmoid) --------
// 128 threads = 4 waves; block tile 64(M)x128(N); wave tile 32x64 = 2x4 WMMA accs.
// K stepped by 32; LDS double-buffered, filled by 6 async b128 copies/thread/tile.
__global__ __launch_bounds__(128)
void gemm_bf16_wmma(const unsigned short* __restrict__ A, const unsigned short* __restrict__ W,
                    const float* __restrict__ bias, float* __restrict__ Y,
                    int M, int N, int Kd, int mode) {
  // per buffer: A 64x32 (2048 bf16, 4KB) | W 128x32 (4096 bf16, 8KB) -> 12KB; x2 buffers
  __shared__ unsigned short smem[2 * 6144];

  const int tid  = threadIdx.x;
  const int lane = tid & 31;
  const int wave = tid >> 5;
  const int wm   = (wave >> 1) * 32;   // 0 or 32
  const int wn   = (wave & 1) * 64;    // 0 or 64
  const int m0   = blockIdx.y * 64;
  const int n0   = blockIdx.x * 128;

  // staging chunks of 8 bf16 (16B). A: 256 chunks -> 2/thread. W: 512 chunks -> 4/thread.
  const int rr = tid >> 2;             // 0..31
  const int q8 = (tid & 3) << 3;       // k sub-offset (elements)
  const int nW0 = min(n0 + rr +  0, N - 1);  // branchless OOB row clamp for W
  const int nW1 = min(n0 + rr + 32, N - 1);
  const int nW2 = min(n0 + rr + 64, N - 1);
  const int nW3 = min(n0 + rr + 96, N - 1);

  const unsigned sbase = lds_offset_of(smem);

  auto issue_tile = [&](int k0, int buf) {
    const unsigned tb = sbase + (unsigned)buf * 12288u;
    const unsigned ao = (unsigned)((rr * 32 + q8) << 1);
    async_copy_b128(tb + ao,         A + ((size_t)(m0 + rr)      * Kd + k0 + q8));
    async_copy_b128(tb + ao + 2048u, A + ((size_t)(m0 + rr + 32) * Kd + k0 + q8));
    const unsigned wb = tb + 4096u + ao;
    async_copy_b128(wb,          W + ((size_t)nW0 * Kd + k0 + q8));
    async_copy_b128(wb + 2048u,  W + ((size_t)nW1 * Kd + k0 + q8));
    async_copy_b128(wb + 4096u,  W + ((size_t)nW2 * Kd + k0 + q8));
    async_copy_b128(wb + 6144u,  W + ((size_t)nW3 * Kd + k0 + q8));
  };

  v8f acc[2][4];
#pragma unroll
  for (int i = 0; i < 2; ++i)
#pragma unroll
    for (int j = 0; j < 4; ++j) acc[i][j] = {};

  const int ntile = Kd >> 5;
  issue_tile(0, 0);
  int buf = 0;
  for (int t = 0; t < ntile; ++t) {
    if (t + 1 < ntile) { issue_tile((t + 1) << 5, buf ^ 1); wait_async_le6(); }
    else               { wait_async_le0(); }
    __syncthreads();

    const unsigned short* As = smem + buf * 6144;
    const unsigned short* Ws = As + 2048;
    const v16bf a0 = frag_from_lds(As, wm + 0,  lane);
    const v16bf a1 = frag_from_lds(As, wm + 16, lane);
    v16bf bfrag[4];
#pragma unroll
    for (int j = 0; j < 4; ++j) bfrag[j] = frag_from_lds(Ws, wn + 16 * j, lane);

#pragma unroll
    for (int j = 0; j < 4; ++j) {
      acc[0][j] = __builtin_amdgcn_wmma_f32_16x16x32_bf16(false, a0, false, bfrag[j], (short)0, acc[0][j], false, false);
      acc[1][j] = __builtin_amdgcn_wmma_f32_16x16x32_bf16(false, a1, false, bfrag[j], (short)0, acc[1][j], false, false);
    }

    __syncthreads();
    buf ^= 1;
  }

  // C/D layout: VGPR r -> M = base + r (+8 for lanes 16-31), N = lane&15.
  const int nn = lane & 15;
  const int mo = (lane >> 4) << 3;
#pragma unroll
  for (int i = 0; i < 2; ++i) {
#pragma unroll
    for (int j = 0; j < 4; ++j) {
#pragma unroll
      for (int r = 0; r < 8; ++r) {
        const int m = m0 + wm + 16 * i + r + mo;
        const int n = n0 + wn + 16 * j + nn;
        if (n < N) {
          float y = acc[i][j][r];
          if (bias) y += bias[n];
          if (mode == 1) y = 1.0f / (1.0f + __expf(-y));
          Y[(size_t)m * N + n] = y;
        }
      }
    }
  }
}

// -------- depthwise causal conv (K=4) + SiLU, channel-fastest --------
__global__ __launch_bounds__(256)
void conv_silu_kernel(const float* __restrict__ X, const float* __restrict__ cw,
                      const float* __restrict__ cb, float* __restrict__ Yo,
                      int S, size_t total) {
  const size_t idx = (size_t)blockIdx.x * blockDim.x + threadIdx.x;
  if (idx >= total) return;
  const int c = (int)(idx % TOT_);
  const int s = (int)((idx / TOT_) % (size_t)S);
  float acc = cb[c];
#pragma unroll
  for (int j = 0; j < KCONV; ++j) {
    const int sp = s - (KCONV - 1) + j;
    if (sp >= 0) acc += cw[c * KCONV + j] * X[idx + ((ptrdiff_t)j - (KCONV - 1)) * TOT_];
  }
  Yo[idx] = acc / (1.0f + __expf(-acc));  // x * sigmoid(x)
}

// -------- wave32 delta recurrence + alpha gate + GroupNorm --------
// One 32-lane wave per (b,h). Lane t owns state rows v=t and v=t+32 (128 VGPRs).
// No LDS, no barriers: broadcasts via __shfl (v_readlane), reductions via __shfl_xor.
__global__ __launch_bounds__(32)
void delta_gn_kernel(const float* __restrict__ qa, const float* __restrict__ ka,
                     const float* __restrict__ va, const float* __restrict__ beta,
                     const float* __restrict__ alpha, const float* __restrict__ st_in,
                     const float* __restrict__ gn_w, const float* __restrict__ gn_b,
                     unsigned short* __restrict__ om, float* __restrict__ st_out, int S) {
  const int b = blockIdx.x >> 4;
  const int h = blockIdx.x & 15;
  const int t = threadIdx.x;                 // 0..31
  const int c0 = h * D_ + t, c1 = c0 + 32;

  float st0[D_], st1[D_];
  {
    const float* s0 = st_in + ((size_t)(b * H_ + h) * D_ + t) * D_;
    const float* s1 = s0 + 32 * D_;
#pragma unroll
    for (int d = 0; d < D_; ++d) { st0[d] = s0[d]; st1[d] = s1[d]; }
  }
  const float gw0 = gn_w[c0], gb0 = gn_b[c0];
  const float gw1 = gn_w[c1], gb1 = gn_b[c1];

  for (int s = 0; s < S; ++s) {
    const size_t row = (size_t)b * S + s;
    const float q0 = qa[row * TOT_ + c0], q1 = qa[row * TOT_ + c1];
    float       k0 = ka[row * TOT_ + c0], k1 = ka[row * TOT_ + c1];
    const float v0 = va[row * TOT_ + c0], v1 = va[row * TOT_ + c1];
    const float a0 = alpha[row * TOT_ + c0], a1 = alpha[row * TOT_ + c1];
    const float bt = beta[row * H_ + h];

    // k <- k / max(||k||2, 1e-12)
    float ss = k0 * k0 + k1 * k1;
#pragma unroll
    for (int m = 16; m >= 1; m >>= 1) ss += __shfl_xor(ss, m, 32);
    const float sc = 1.0f / fmaxf(sqrtf(ss), 1e-12f);
    k0 *= sc; k1 *= sc;

    // retrieval = state . k
    float r0 = 0.0f, r1 = 0.0f;
#pragma unroll
    for (int d = 0; d < 32; ++d) {
      const float kd = __shfl(k0, d, 32);
      r0 += st0[d] * kd; r1 += st1[d] * kd;
    }
#pragma unroll
    for (int d = 0; d < 32; ++d) {
      const float kd = __shfl(k1, d, 32);
      r0 += st0[d + 32] * kd; r1 += st1[d + 32] * kd;
    }
    const float co0 = bt * (v0 - r0), co1 = bt * (v1 - r1);

    // state += beta*delta (x) k ; o = state . q
    float o0 = 0.0f, o1 = 0.0f;
#pragma unroll
    for (int d = 0; d < 32; ++d) {
      const float kd = __shfl(k0, d, 32), qd = __shfl(q0, d, 32);
      st0[d] += co0 * kd; st1[d] += co1 * kd;
      o0 += st0[d] * qd;  o1 += st1[d] * qd;
    }
#pragma unroll
    for (int d = 0; d < 32; ++d) {
      const float kd = __shfl(k1, d, 32), qd = __shfl(q1, d, 32);
      st0[d + 32] += co0 * kd; st1[d + 32] += co1 * kd;
      o0 += st0[d + 32] * qd;  o1 += st1[d + 32] * qd;
    }

    const float ov0 = o0 * a0, ov1 = o1 * a1;

    // GroupNorm over the head's 64 channels
    float sm = ov0 + ov1, sq = ov0 * ov0 + ov1 * ov1;
#pragma unroll
    for (int m = 16; m >= 1; m >>= 1) {
      sm += __shfl_xor(sm, m, 32);
      sq += __shfl_xor(sq, m, 32);
    }
    const float mu  = sm * (1.0f / D_);
    const float var = sq * (1.0f / D_) - mu * mu;
    const float inv = rsqrtf(var + 1e-5f);
    om[row * TOT_ + c0] = f2bf((ov0 - mu) * inv * gw0 + gb0);
    om[row * TOT_ + c1] = f2bf((ov1 - mu) * inv * gw1 + gb1);
  }

  float* so0 = st_out + ((size_t)(b * H_ + h) * D_ + t) * D_;
  float* so1 = so0 + 32 * D_;
#pragma unroll
  for (int d = 0; d < D_; ++d) { so0[d] = st0[d]; so1[d] = st1[d]; }
}

extern "C" void kernel_launch(void* const* d_in, const int* in_sizes, int n_in,
                              void* d_out, int out_size, void* d_ws, size_t ws_size,
                              hipStream_t stream) {
  const float* x       = (const float*)d_in[0];
  const float* state   = (const float*)d_in[1];
  const float* q_w     = (const float*)d_in[2];
  const float* k_w     = (const float*)d_in[3];
  const float* v_w     = (const float*)d_in[4];
  const float* q_cw    = (const float*)d_in[5];
  const float* q_cb    = (const float*)d_in[6];
  const float* k_cw    = (const float*)d_in[7];
  const float* k_cb    = (const float*)d_in[8];
  const float* v_cw    = (const float*)d_in[9];
  const float* v_cb    = (const float*)d_in[10];
  const float* beta_w  = (const float*)d_in[11];
  const float* beta_b  = (const float*)d_in[12];
  const float* alpha_w = (const float*)d_in[13];
  const float* alpha_b = (const float*)d_in[14];
  const float* o_w     = (const float*)d_in[15];
  const float* gn_w    = (const float*)d_in[16];
  const float* gn_b    = (const float*)d_in[17];

  const int B  = in_sizes[1] / (H_ * D_ * D_);  // state: [B,H,D,D]
  const int Bs = in_sizes[0] / HID_;            // B*S
  const int S  = Bs / B;
  const size_t MT = (size_t)Bs * TOT_;

  // workspace carve-out (256B aligned)
  size_t off = 0;
  auto alloc = [&](size_t bytes) -> void* {
    void* p = (char*)d_ws + off;
    off += (bytes + 255) & ~(size_t)255;
    return p;
  };
  unsigned short* xb  = (unsigned short*)alloc(MT * 2);            // bf16(x)
  unsigned short* wqb = (unsigned short*)alloc((size_t)TOT_ * HID_ * 2);
  unsigned short* wkb = (unsigned short*)alloc((size_t)TOT_ * HID_ * 2);
  unsigned short* wvb = (unsigned short*)alloc((size_t)TOT_ * HID_ * 2);
  unsigned short* wab = (unsigned short*)alloc((size_t)TOT_ * HID_ * 2);
  unsigned short* wbb = (unsigned short*)alloc((size_t)H_ * HID_ * 2);
  unsigned short* wob = (unsigned short*)alloc((size_t)HID_ * TOT_ * 2);
  unsigned short* omb = (unsigned short*)alloc(MT * 2);            // bf16 recurrence output
  float* qr = (float*)alloc(MT * 4);
  float* kr = (float*)alloc(MT * 4);
  float* vr = (float*)alloc(MT * 4);
  float* qa = (float*)alloc(MT * 4);
  float* ka = (float*)alloc(MT * 4);
  float* va = (float*)alloc(MT * 4);
  float* al = (float*)alloc(MT * 4);
  float* bt = (float*)alloc((size_t)Bs * H_ * 4);

  float* out_y  = (float*)d_out;                 // [B,S,HID]
  float* out_st = out_y + (size_t)Bs * HID_;     // [B,H,D,D]

  // 1. f32 -> bf16 conversions (one-time)
  auto cvt = [&](const float* s, unsigned short* d, size_t n) {
    const int blocks = (int)((n / 4 + 255) / 256);
    cvt_bf16_kernel<<<blocks, 256, 0, stream>>>(s, d, n);
  };
  cvt(x,       xb,  MT);
  cvt(q_w,     wqb, (size_t)TOT_ * HID_);
  cvt(k_w,     wkb, (size_t)TOT_ * HID_);
  cvt(v_w,     wvb, (size_t)TOT_ * HID_);
  cvt(alpha_w, wab, (size_t)TOT_ * HID_);
  cvt(beta_w,  wbb, (size_t)H_ * HID_);
  cvt(o_w,     wob, (size_t)HID_ * TOT_);

  // 2. projections (bf16 WMMA, async-LDS double-buffered)
  const dim3 blk(128);
  const dim3 grid_full((TOT_ + 127) / 128, (Bs + 63) / 64);
  const dim3 grid_beta(1, (Bs + 63) / 64);
  gemm_bf16_wmma<<<grid_full, blk, 0, stream>>>(xb, wqb, nullptr, qr, Bs, TOT_, HID_, 0);
  gemm_bf16_wmma<<<grid_full, blk, 0, stream>>>(xb, wkb, nullptr, kr, Bs, TOT_, HID_, 0);
  gemm_bf16_wmma<<<grid_full, blk, 0, stream>>>(xb, wvb, nullptr, vr, Bs, TOT_, HID_, 0);
  gemm_bf16_wmma<<<grid_full, blk, 0, stream>>>(xb, wab, alpha_b, al, Bs, TOT_, HID_, 1);
  gemm_bf16_wmma<<<grid_beta, blk, 0, stream>>>(xb, wbb, beta_b,  bt, Bs, H_,   HID_, 1);

  // 3. causal depthwise conv + SiLU
  const size_t total = MT;
  const int cblocks = (int)((total + 255) / 256);
  conv_silu_kernel<<<cblocks, 256, 0, stream>>>(qr, q_cw, q_cb, qa, S, total);
  conv_silu_kernel<<<cblocks, 256, 0, stream>>>(kr, k_cw, k_cb, ka, S, total);
  conv_silu_kernel<<<cblocks, 256, 0, stream>>>(vr, v_cw, v_cb, va, S, total);

  // 4. sequential delta recurrence (+gate +GroupNorm), final state to output tail
  delta_gn_kernel<<<B * H_, 32, 0, stream>>>(qa, ka, va, bt, al, state, gn_w, gn_b,
                                             omb, out_st, S);

  // 5. output projection
  const dim3 grid_out((HID_ + 127) / 128, (Bs + 63) / 64);
  gemm_bf16_wmma<<<grid_out, blk, 0, stream>>>(omb, wob, nullptr, out_y, Bs, HID_, TOT_, 0);
}
